// NCO_63436666962542
// MI455X (gfx1250) — matrix operand order repaired and checked by reading, not compiled, not run on hardware
//
#include <hip/hip_runtime.h>
#include <hip/hip_bf16.h>

#define N_SAMPLES 64
#define N_ASSETS  256
#define N_CLUST   8
#define N_INIT    10
#define MAX_ITER  30

typedef __attribute__((ext_vector_type(8)))  float        v8f;
typedef __attribute__((ext_vector_type(16))) __bf16       v16bf;
typedef __attribute__((ext_vector_type(8)))  unsigned int v8u;
typedef __attribute__((ext_vector_type(4)))  unsigned int v4u;
typedef __attribute__((ext_vector_type(4)))  int          v4i;
typedef __attribute__((ext_vector_type(8)))  int          v8i;

__device__ __forceinline__ unsigned short f2bf(float f) {
  unsigned int u = __float_as_uint(f);
  u += 0x7FFFu + ((u >> 16) & 1u);          // round-to-nearest-even
  return (unsigned short)(u >> 16);
}
__device__ __forceinline__ float bf2f(unsigned short h) {
  return __uint_as_float(((unsigned int)h) << 16);
}

// ---------------------------------------------------------------------------
// Kernel 1: corr = cov / (sd sd^T) stored as bf16; xnorm = ||corr row||^2
// ---------------------------------------------------------------------------
__global__ __launch_bounds__(256) void nco_corr_kernel(
    const float* __restrict__ cov,
    unsigned short* __restrict__ corr_bf,
    float* __restrict__ xnorm) {
  const int s = blockIdx.x, j = threadIdx.x;
  const float* C = cov + (size_t)s * N_ASSETS * N_ASSETS;
  unsigned short* G = corr_bf + (size_t)s * N_ASSETS * N_ASSETS;
  __shared__ float invsd[N_ASSETS];
  invsd[j] = rsqrtf(C[j * N_ASSETS + j]);
  __syncthreads();
  const float isj = invsd[j];
  for (int i = 0; i < N_ASSETS; ++i)
    G[i * N_ASSETS + j] = f2bf(C[i * N_ASSETS + j] * invsd[i] * isj);
  __syncthreads();
  float acc = 0.f;
  for (int q = 0; q < N_ASSETS; ++q) {
    float v = bf2f(G[j * N_ASSETS + q]);
    acc += v * v;
  }
  xnorm[s * N_ASSETS + j] = acc;
}

// ---------------------------------------------------------------------------
// Kernel 2: Lloyd's k-means, one block per (sample, init). 8 waves.
// corr tile staged Global->LDS by the Tensor Data Mover (one TENSOR_LOAD_TO_LDS,
// TENSORcnt-tracked); distance dot-products via v_wmma_f32_16x16x32_bf16.
// LDS layout (dynamic, 167424 B):
//   corr_l  [256][256] bf16 @      0  (131072)
//   cent_bf [ 16][256] bf16 @ 131072  (  8192)   clusters 8..15 zero
//   cent_f  [  8][256] f32  @ 139264  (  8192)
//   dots    [256][ 16] f32  @ 147456  ( 16384)
//   xn      [256]      f32  @ 163840  (  1024)
//   cn      [ 16]      f32  @ 164864  (    64)
//   asg     [256]      i32  @ 164928  (  1024)
//   counts  [ 16]      i32  @ 165952  (    64)
//   red     [256]      f32  @ 166016  (  1024)
//   iidx    [  8]      i32  @ 167040  (    32)
// ---------------------------------------------------------------------------
__global__ __launch_bounds__(256) void nco_kmeans_kernel(
    const unsigned short* __restrict__ corr_bf,
    const float* __restrict__ xnorm_g,
    int* __restrict__ assign_all,
    float* __restrict__ inertia_all) {
  const int s = blockIdx.x, init = blockIdx.y, t = threadIdx.x;
  const int lane = t & 31, wave = t >> 5;
  extern __shared__ char smem[];
  unsigned short* corr_l  = (unsigned short*)(smem);
  unsigned short* cent_bf = (unsigned short*)(smem + 131072);
  float* cent_f  = (float*)(smem + 139264);
  float* dots    = (float*)(smem + 147456);
  float* xn      = (float*)(smem + 163840);
  float* cn      = (float*)(smem + 164864);
  int*   asg     = (int*)  (smem + 164928);
  int*   counts  = (int*)  (smem + 165952);
  float* red     = (float*)(smem + 166016);
  int*   iidx    = (int*)  (smem + 167040);

  const unsigned short* G = corr_bf + (size_t)s * N_ASSETS * N_ASSETS;

  // --- TDM: one tensor_load_to_lds DMAs the whole 256x256 bf16 tile ---
  // D# group0: count=1 | lds_addr | global_addr[56:0] | type=2 (bits 127:126)
  // D# group1: data_size=1 (2B); tensor_dim0/1=256; tile_dim0/1=256;
  //            tensor_dim0_stride=256. Groups 2/3 zero (2-D tensor).
  if (wave == 0) {
    unsigned long long ga = (unsigned long long)G;
    unsigned lds_addr = (unsigned)(unsigned long long)corr_l;  // flat low 32b = LDS byte offset
    v4u g0 = { 1u,                                        // count=1, is_restore=0
               lds_addr,                                  // bits 63:32
               (unsigned)(ga & 0xFFFFFFFFu),              // global_addr lo
               (unsigned)((ga >> 32) & 0x01FFFFFFu) | 0x80000000u };  // addr[56:32] | type=2
    v8i g1 = { (int)0x00010000,       // workgroup_mask=0, data_size=1 (2B)
               (int)(256u << 16),     // tensor_dim0 = 256 (low 16 bits at [31:16])
               (int)(256u << 16),     // tensor_dim0 hi = 0 | tensor_dim1 = 256
               (int)(256u << 16),     // tensor_dim1 hi = 0 | tile_dim0 = 256
               (int)256,              // tile_dim1 = 256, tile_dim2 = 0
               (int)256,              // tensor_dim0_stride = 256 (lo 32)
               0,                     // stride0 hi = 0, tensor_dim1_stride lo16 = 0
               0 };
    v4i gz = { 0, 0, 0, 0 };
#if defined(__clang_major__) && (__clang_major__ >= 23)
    v8i gz8 = { 0, 0, 0, 0, 0, 0, 0, 0 };
    __builtin_amdgcn_tensor_load_to_lds(g0, g1, gz, gz, gz8, 0);
#else
    __builtin_amdgcn_tensor_load_to_lds(g0, g1, gz, gz, 0);
#endif
    __builtin_amdgcn_s_wait_tensorcnt(0);
  }

  xn[t] = xnorm_g[s * N_ASSETS + t];

  // 8 distinct pseudo-random starting rows (deterministic LCG keyed by (s,init))
  if (t == 0) {
    unsigned rng = (unsigned)(s * 2654435761u) ^ (unsigned)(init * 0x9E3779B9u) ^ 42u;
    unsigned used[8] = {0, 0, 0, 0, 0, 0, 0, 0};
    for (int c = 0; c < N_CLUST; ++c) {
      int idx;
      do {
        rng = rng * 1664525u + 1013904223u;
        idx = (int)((rng >> 10) & 255u);
      } while (used[idx >> 5] & (1u << (idx & 31)));
      used[idx >> 5] |= (1u << (idx & 31));
      iidx[c] = idx;
    }
  }
  __syncthreads();   // TDM tile + iidx visible to all waves
  for (int c = 0; c < N_CLUST; ++c) {
    unsigned short h = corr_l[iidx[c] * N_ASSETS + t];
    cent_bf[c * N_ASSETS + t] = h;
    cent_f [c * N_ASSETS + t] = bf2f(h);
  }
  for (int c = N_CLUST; c < 16; ++c) cent_bf[c * N_ASSETS + t] = 0;
  __syncthreads();

  for (int iter = 0; iter <= MAX_ITER; ++iter) {
    // center norms (bf16-rounded, consistent with the WMMA dot term)
    if (t < N_CLUST) {
      float a = 0.f;
      for (int q = 0; q < N_ASSETS; ++q) {
        float v = bf2f(cent_bf[t * N_ASSETS + q]);
        a += v * v;
      }
      cn[t] = a;
    }
    __syncthreads();

    // dot(x_i, c_c): 256x256 @ 256x16 bf16 GEMM via WMMA. Wave w owns row
    // tiles w and w+8; K=256 covered by 8 steps of 32.
    const int khalf = lane >> 4;    // 0: lanes 0-15, 1: lanes 16-31
    const int rsel  = lane & 15;
    #pragma unroll
    for (int mrep = 0; mrep < 2; ++mrep) {
      const int mt  = wave + 8 * mrep;
      const int row = mt * 16 + rsel;
      v8f acc = {0.f, 0.f, 0.f, 0.f, 0.f, 0.f, 0.f, 0.f};
      #pragma unroll
      for (int kk = 0; kk < 8; ++kk) {
        const int kb = kk * 32;
        // A fragment (16x32 bf16): lane<16 holds K {0..7,16..23}, lane>=16
        // holds K {8..15,24..31} of row `row` (2 bf16 per VGPR).
        uint4 alo = *(const uint4*)(corr_l + row * N_ASSETS + kb + khalf * 8);
        uint4 ahi = *(const uint4*)(corr_l + row * N_ASSETS + kb + 16 + khalf * 8);
        v8u aw = {alo.x, alo.y, alo.z, alo.w, ahi.x, ahi.y, ahi.z, ahi.w};
        // B fragment (32x16 bf16): lane n holds column n, K {0..7,16..23} (lane<16) or {8..15,24..31}
        const unsigned short* bp = cent_bf + rsel * N_ASSETS + kb + khalf * 8;
        uint4 blo = *(const uint4*)(bp);
        uint4 bhi = *(const uint4*)(bp + 16);
        v8u bw = {blo.x, blo.y, blo.z, blo.w, bhi.x, bhi.y, bhi.z, bhi.w};
        acc = __builtin_amdgcn_wmma_f32_16x16x32_bf16(
            false, __builtin_bit_cast(v16bf, aw),
            false, __builtin_bit_cast(v16bf, bw),
            (short)0, acc, false, false);
      }
      // D layout: VGPR v -> M = v + 8*khalf, N = rsel
      #pragma unroll
      for (int v = 0; v < 8; ++v)
        dots[(mt * 16 + v + 8 * khalf) * 16 + rsel] = acc[v];
    }
    __syncthreads();

    // assignment: d2 = ||x||^2 + ||c||^2 - 2 x.c
    {
      float best = 3.4e38f;
      int bc = 0;
      const float xv = xn[t];
      #pragma unroll
      for (int c = 0; c < N_CLUST; ++c) {
        float d2 = xv + cn[c] - 2.f * dots[t * 16 + c];
        if (d2 < best) { best = d2; bc = c; }
      }
      asg[t] = bc;
      red[t] = best;
    }
    __syncthreads();
    if (iter == MAX_ITER) break;   // final pass: assignment + inertia only

    if (t < 16) counts[t] = 0;
    __syncthreads();
    atomicAdd(&counts[asg[t]], 1);
    __syncthreads();

    // new centers; thread t owns dimension t
    {
      float a[8] = {0.f, 0.f, 0.f, 0.f, 0.f, 0.f, 0.f, 0.f};
      for (int p = 0; p < N_ASSETS; ++p) {
        float v = bf2f(corr_l[p * N_ASSETS + t]);
        int c = asg[p];
        #pragma unroll
        for (int cc = 0; cc < 8; ++cc) a[cc] += (cc == c) ? v : 0.f;
      }
      #pragma unroll
      for (int c = 0; c < N_CLUST; ++c) {
        float nc = (counts[c] > 0) ? (a[c] / (float)counts[c])
                                   : cent_f[c * N_ASSETS + t];
        cent_f [c * N_ASSETS + t] = nc;
        cent_bf[c * N_ASSETS + t] = f2bf(nc);
      }
    }
    __syncthreads();
  }

  // inertia = sum of min d2
  for (int off = 128; off > 0; off >>= 1) {
    if (t < off) red[t] += red[t + off];
    __syncthreads();
  }
  const int slot = s * N_INIT + init;
  if (t == 0) inertia_all[slot] = red[0];
  assign_all[slot * N_ASSETS + t] = asg[t];
}

// ---------------------------------------------------------------------------
// Kernel 3: pick best init per sample
// ---------------------------------------------------------------------------
__global__ __launch_bounds__(256) void nco_best_kernel(
    const float* __restrict__ inertia_all,
    const int* __restrict__ assign_all,
    int* __restrict__ best_assign) {
  const int s = blockIdx.x, t = threadIdx.x;
  __shared__ int bsh;
  if (t == 0) {
    float bi = inertia_all[s * N_INIT];
    int b = 0;
    for (int i = 1; i < N_INIT; ++i) {
      float v = inertia_all[s * N_INIT + i];
      if (v < bi) { bi = v; b = i; }
    }
    bsh = b;
  }
  __syncthreads();
  best_assign[s * N_ASSETS + t] = assign_all[(s * N_INIT + bsh) * N_ASSETS + t];
}

// ---------------------------------------------------------------------------
// Kernel 4: masked Markowitz solve, one block per (sample, cluster).
// Augmented 256x257 f32 matrix fully LDS-resident (needs CDNA5's 320KB LDS).
// Column-parallel LU (no pivoting: SPD-block (+) identity), parallel back-sub.
// Dynamic LDS 267264 B: A@0 (263168), fcol@263168, wv@264192, asg@265216,
// red@266240 (1024 each).
// ---------------------------------------------------------------------------
__global__ __launch_bounds__(256) void nco_solve_kernel(
    const float* __restrict__ cov,
    const float* __restrict__ rets,
    const int* __restrict__ best_assign,
    float* __restrict__ w_cols) {
  const int s = blockIdx.x >> 3, c = blockIdx.x & 7, t = threadIdx.x;
  extern __shared__ char smem[];
  float* A    = (float*)(smem);
  float* fcol = (float*)(smem + 263168);
  float* wv   = (float*)(smem + 264192);
  int*   asg  = (int*)  (smem + 265216);
  float* red  = (float*)(smem + 266240);
  const float* C = cov + (size_t)s * N_ASSETS * N_ASSETS;

  asg[t] = best_assign[s * N_ASSETS + t];
  __syncthreads();
  const int mj = (asg[t] == c);
  for (int i = 0; i < N_ASSETS; ++i) {
    int mi = (asg[i] == c);
    A[i * 257 + t] = (mi && mj) ? C[i * N_ASSETS + t] : ((i == t) ? 1.f : 0.f);
  }
  A[t * 257 + 256] = mj ? rets[s * N_ASSETS + t] : 0.f;
  __syncthreads();

  for (int k = 0; k < N_ASSETS - 1; ++k) {
    const float inv = 1.f / A[k * 257 + k];
    if (t > k) fcol[t] = A[t * 257 + k] * inv;
    __syncthreads();
    const float akj = A[k * 257 + t];
    const float akb = A[k * 257 + 256];
    for (int i = k + 1; i < N_ASSETS; ++i) {
      const float f = fcol[i];
      A[i * 257 + t] -= f * akj;
      if (t == 255) A[i * 257 + 256] -= f * akb;  // RHS column
    }
    __syncthreads();
  }
  for (int k = N_ASSETS - 1; k >= 0; --k) {
    if (t == k) wv[k] = A[k * 257 + 256] / A[k * 257 + k];
    __syncthreads();
    if (t < k) A[t * 257 + 256] -= A[t * 257 + k] * wv[k];
    __syncthreads();
  }
  red[t] = wv[t];
  __syncthreads();
  for (int off = 128; off > 0; off >>= 1) {
    if (t < off) red[t] += red[t + off];
    __syncthreads();
  }
  const float denom = red[0];
  w_cols[((size_t)s * N_CLUST + c) * N_ASSETS + t] =
      (denom != 0.f) ? wv[t] / denom : 0.f;
}

// ---------------------------------------------------------------------------
// Kernel 5: inter-cluster 8x8 system + final combine, one block per sample.
// ---------------------------------------------------------------------------
__global__ __launch_bounds__(256) void nco_combine_kernel(
    const float* __restrict__ cov,
    const float* __restrict__ rets,
    const float* __restrict__ w_cols,
    float* __restrict__ out) {
  const int s = blockIdx.x, t = threadIdx.x;
  __shared__ float wl[N_ASSETS * 8];
  __shared__ float tl[N_ASSETS * 8];
  __shared__ float ic[64], ir[8], wu[8], winter[8];
  const float* C = cov + (size_t)s * N_ASSETS * N_ASSETS;

  #pragma unroll
  for (int c = 0; c < 8; ++c)
    wl[t * 8 + c] = w_cols[((size_t)s * 8 + c) * N_ASSETS + t];
  __syncthreads();

  // tl = cov @ w_intra (256x8); cov rows are L2-hot, prefetch ahead
  {
    float a[8] = {0.f, 0.f, 0.f, 0.f, 0.f, 0.f, 0.f, 0.f};
    for (int j = 0; j < N_ASSETS; ++j) {
      if ((j & 31) == 0 && j + 32 < N_ASSETS)
        __builtin_prefetch(&C[t * N_ASSETS + j + 32], 0, 0);
      const float cv = C[t * N_ASSETS + j];
      #pragma unroll
      for (int c = 0; c < 8; ++c) a[c] += cv * wl[j * 8 + c];
    }
    #pragma unroll
    for (int c = 0; c < 8; ++c) tl[t * 8 + c] = a[c];
  }
  __syncthreads();

  if (t < 64) {  // inter_cov = w_intra^T @ tl
    const int c1 = t >> 3, c2 = t & 7;
    float a = 0.f;
    for (int i = 0; i < N_ASSETS; ++i) a += wl[i * 8 + c1] * tl[i * 8 + c2];
    ic[t] = a;
  }
  if (t >= 64 && t < 72) {  // inter_rets = w_intra^T @ r
    const int c = t - 64;
    float a = 0.f;
    for (int i = 0; i < N_ASSETS; ++i) a += wl[i * 8 + c] * rets[s * N_ASSETS + i];
    ir[c] = a;
  }
  __syncthreads();

  if (t == 0) {  // 8x8 solve with partial pivoting
    for (int k = 0; k < 8; ++k) {
      int p = k;
      float mx = fabsf(ic[k * 8 + k]);
      for (int i = k + 1; i < 8; ++i) {
        float v = fabsf(ic[i * 8 + k]);
        if (v > mx) { mx = v; p = i; }
      }
      if (p != k) {
        for (int j = 0; j < 8; ++j) {
          float tmp = ic[k * 8 + j]; ic[k * 8 + j] = ic[p * 8 + j]; ic[p * 8 + j] = tmp;
        }
        float tr = ir[k]; ir[k] = ir[p]; ir[p] = tr;
      }
      const float inv = 1.f / ic[k * 8 + k];
      for (int i = k + 1; i < 8; ++i) {
        const float f = ic[i * 8 + k] * inv;
        for (int j = k; j < 8; ++j) ic[i * 8 + j] -= f * ic[k * 8 + j];
        ir[i] -= f * ir[k];
      }
    }
    for (int k = 7; k >= 0; --k) {
      float a = ir[k];
      for (int j = k + 1; j < 8; ++j) a -= ic[k * 8 + j] * wu[j];
      wu[k] = a / ic[k * 8 + k];
    }
    float ssum = 0.f;
    for (int k = 0; k < 8; ++k) ssum += wu[k];
    for (int k = 0; k < 8; ++k) winter[k] = wu[k] / ssum;
  }
  __syncthreads();

  {
    float a = 0.f;
    #pragma unroll
    for (int c = 0; c < 8; ++c) a += wl[t * 8 + c] * winter[c];
    out[s * N_ASSETS + t] = a;
  }
}

// ---------------------------------------------------------------------------
// Workspace layout (bytes):
//   corr_bf     @        0   64*256*256*2 = 8388608
//   xnorm       @  8388608   64*256*4     =   65536
//   assign_all  @  8454144   64*10*256*4  =  655360
//   inertia_all @  9109504   64*10*4      =    2560 (padded to 4096)
//   best_assign @  9113600   64*256*4     =   65536
//   w_cols      @  9179136   64*8*256*4   =  524288
//   total 9703424 bytes
// ---------------------------------------------------------------------------
extern "C" void kernel_launch(void* const* d_in, const int* in_sizes, int n_in,
                              void* d_out, int out_size, void* d_ws, size_t ws_size,
                              hipStream_t stream) {
  (void)in_sizes; (void)n_in; (void)out_size; (void)ws_size;
  const float* cov  = (const float*)d_in[0];
  const float* rets = (const float*)d_in[1];
  float* out = (float*)d_out;
  char* ws = (char*)d_ws;

  unsigned short* corr_bf  = (unsigned short*)(ws);
  float* xnorm             = (float*)(ws + 8388608);
  int*   assign_all        = (int*)  (ws + 8454144);
  float* inertia_all       = (float*)(ws + 9109504);
  int*   best_assign       = (int*)  (ws + 9113600);
  float* w_cols            = (float*)(ws + 9179136);

  hipFuncSetAttribute((const void*)nco_kmeans_kernel,
                      hipFuncAttributeMaxDynamicSharedMemorySize, 167424);
  hipFuncSetAttribute((const void*)nco_solve_kernel,
                      hipFuncAttributeMaxDynamicSharedMemorySize, 267264);

  nco_corr_kernel<<<N_SAMPLES, N_ASSETS, 0, stream>>>(cov, corr_bf, xnorm);
  nco_kmeans_kernel<<<dim3(N_SAMPLES, N_INIT), N_ASSETS, 167424, stream>>>(
      corr_bf, xnorm, assign_all, inertia_all);
  nco_best_kernel<<<N_SAMPLES, N_ASSETS, 0, stream>>>(
      inertia_all, assign_all, best_assign);
  nco_solve_kernel<<<N_SAMPLES * N_CLUST, N_ASSETS, 267264, stream>>>(
      cov, rets, best_assign, w_cols);
  nco_combine_kernel<<<N_SAMPLES, N_ASSETS, 0, stream>>>(
      cov, rets, w_cols, out);
}
